// CustomInteractionBlock_29231547416903
// MI455X (gfx1250) — compile-verified, matrix operand order checked by reading, not run on hardware
//
#include <hip/hip_runtime.h>
#include <hip/hip_bf16.h>
#include <stdint.h>

// ---------------- problem constants ----------------
#define N_NODES   8000
#define N_EDGES   80000
#define HIDDEN    64
#define DIM_IN    80            // 32 + 3*16
#define WNUMEL    3328
#define NCHUNK    208           // 3328 / 16
#define NGROUP    52            // 208 / 4 chunks per staging group
#define M_CH      96            // message channels: 32 scalars + 16 gates + 48 vec

#define C_PI      3.14159265358979f
#define R_MAX_F   5.0f
#define SILU_C    1.18551f      // analytic 1/sqrt(E[silu(z)^2]), z~N(0,1)
#define SIG_C     1.30545f      // analytic 1/sqrt(E[sigmoid(z)^2])
#define A_SC      0.14433757f   // 1/sqrt(48)
#define A_VE      0.21650635f   // sqrt(3/64)
#define INV3      0.57735027f   // 1/sqrt(3)
#define S2C       0.70710678f   // 1/sqrt(2)
#define S6C       0.40824829f   // 1/sqrt(6)
#define R5C       0.44721360f   // 1/sqrt(5)
#define INV_SQRT8  0.35355339f
#define INV_SQRT32 0.17677670f

typedef __attribute__((ext_vector_type(16))) _Float16 v16h;
typedef __attribute__((ext_vector_type(8)))  _Float16 v8h;
typedef __attribute__((ext_vector_type(8)))  float    v8f;
typedef __attribute__((ext_vector_type(4)))  uint32_t u32x4;
typedef __attribute__((ext_vector_type(8)))  int      i32x8;
typedef __attribute__((ext_vector_type(4)))  int      i32x4;

// TDM async staging available? (this toolchain: 6-arg builtin, clang-23 form)
#if defined(__AMDGCN__) && __has_builtin(__builtin_amdgcn_tensor_load_to_lds) && \
    __has_builtin(__builtin_amdgcn_s_wait_tensorcnt)
#define USE_TDM 1
#else
#define USE_TDM 0
#endif

// ws layout
#define PACK_DWORDS (NCHUNK * 2 * 32 * 8)        // 106496 dwords = 425984 B
#define MSUM_FLOATS (N_NODES * M_CH)             // 768000
#define CNT_FLOATS  (N_NODES)

// ---------------------------------------------------------------------------
// Kernel 1: pack fc_w2 (f32 [64][3328]) into f16 B-operand layout.
// packed[((chunk*2+kv)*32 + lane)*8 + t] = dword with f16 pair (K0, K0+1).
// B operand layout (v_wmma_f32_16x16x32_f16, 32x16 B): lane<16 -> N=lane,
// vgpr t<4: K=2t,2t+1 ; t>=4: K=2t+8 ; lanes>=16 add +8 to K, N=lane-16.
// kv selects K offset 0 / 32 (two WMMAs cover K=64).
// ---------------------------------------------------------------------------
__global__ void pack_w2_kernel(const float* __restrict__ fc_w2,
                               uint32_t* __restrict__ packed) {
    int id = blockIdx.x * 256 + threadIdx.x;
    if (id >= PACK_DWORDS) return;
    int t    = id & 7;
    int lane = (id >> 3) & 31;
    int kv   = (id >> 8) & 1;
    int c    = id >> 9;
    int n    = c * 16 + (lane & 15);
    int lh   = lane >> 4;
    int k0   = 32 * kv + ((t < 4) ? (2 * t) : (2 * t + 8)) + 8 * lh;
    union { _Float16 f[2]; uint32_t u; } pk;
    pk.f[0] = (_Float16)fc_w2[k0 * WNUMEL + n];
    pk.f[1] = (_Float16)fc_w2[(k0 + 1) * WNUMEL + n];
    packed[id] = pk.u;
}

// ---------------------------------------------------------------------------
// Kernel 2: zero m_sum + cnt
// ---------------------------------------------------------------------------
__global__ void zero_kernel(float* __restrict__ p, int n) {
    int i = blockIdx.x * 256 + threadIdx.x;
    if (i < n) p[i] = 0.0f;
}

// ---------------------------------------------------------------------------
// Kernel 3: fused edge kernel.
//   4 waves/block, 2 tiles (of 16 edges) per wave -> 128 edges per block.
//   Packed fc_w2 staged group-by-group (4 chunks = 8 KB) in double-buffered
//   LDS shared by all 4 waves; each staged operand feeds both tiles of each
//   wave -> 8x reduction in L2 B-operand traffic (2.1 GB -> 266 MB).
//   Staging is done by the Tensor Data Mover (tensor_load_to_lds, TENSORcnt)
//   so no VGPRs are tied up in the copy (fallback: immediate load+ds_store).
//   Seven statically-dispatched block loops keep all accumulators in VGPRs.
// ---------------------------------------------------------------------------
struct __align__(16) WaveShm {
    _Float16 hbuf[16][64];   // 2048 B  h in f16 (x 0.125 x SILU_C folded)
    float    rb[16][8];      // 512 B
    float    fA[32][16];     // [u][e]  A_SC*a0*sj          (blocks A,C)
    float    fB[16][16];     // [u][e]  A_SC*INV3*vd        (blocks B,D)
    float    fE[32][16];     // [u][e]  A_VE*INV3*sj        (block E)
    float    fF[16][3][16];  // [u][j][e] A_VE*INV3*a0*vj   (block F)
    float    fG[16][3][16];  // [u][j][e] A_VE*(vj . t)     (block G)
    float    a1v[3][16];     // [j][e]  raw a1              (E epilogue)
    int      dstIdx[16];
};                           // 14080 B

#define TILES_PER_WAVE 2
#define WAVES_PER_BLK  4
#define TILES_PER_BLK  (TILES_PER_WAVE * WAVES_PER_BLK)   // 8
#define GRP_DWORDS     2048                                // 4 chunks * 512
#define DYN_LDS_BYTES  (TILES_PER_BLK * (int)sizeof(WaveShm) + 2 * GRP_DWORDS * 4)

struct Acc { float sc0[8], sc1[8], gt[8], te[8], vc0[8], vc1[8], vc2[8]; };

__global__ void __launch_bounds__(128)
edge_kernel(const float* __restrict__ x,
            const float* __restrict__ edge_attr,
            const float* __restrict__ edge_length,
            const int*   __restrict__ edge_src,
            const int*   __restrict__ edge_dst,
            const float* __restrict__ fc_w1,
            const uint32_t* __restrict__ wp,   // packed fc_w2
            float* __restrict__ m_sum,
            float* __restrict__ cnt) {
    extern __shared__ char dynlds[];
    WaveShm*  tileShm = (WaveShm*)dynlds;
    uint32_t* bstage  = (uint32_t*)(dynlds + TILES_PER_BLK * sizeof(WaveShm));

    const int tid  = threadIdx.x;
    const int lane = tid & 31;
    const int wave = tid >> 5;
    WaveShm* TS[TILES_PER_WAVE];
    int eBase[TILES_PER_WAVE];
    #pragma unroll
    for (int t = 0; t < TILES_PER_WAVE; ++t) {
        TS[t] = &tileShm[wave * TILES_PER_WAVE + t];
        eBase[t] = (blockIdx.x * TILES_PER_BLK + wave * TILES_PER_WAVE + t) * 16;
    }

    // ---- staging: group g = chunks [4g, 4g+4) = 8 KB, double buffered ----
#if USE_TDM
    // TDM 1-D descriptor: 1024 x 8-byte elements, contiguous.
    auto tdmStage = [&](int g) {
        uint32_t* buf = bstage + (g & 1) * GRP_DWORDS;
        uint64_t ga = (uint64_t)(uintptr_t)(wp + (size_t)g * GRP_DWORDS);
        uint32_t la = (uint32_t)(uintptr_t)buf;   // generic LDS ptr low 32 = LDS addr
        u32x4 g0;
        g0[0] = 1u;                                   // count=1, user mode
        g0[1] = la;                                   // lds_addr
        g0[2] = (uint32_t)ga;                         // global_addr[31:0]
        g0[3] = (uint32_t)(ga >> 32) | (2u << 30);    // global_addr[56:32], type=2
        i32x8 g1;
        g1[0] = (int)(3u << 16);                      // wg_mask=0, data_size=3 (8B)
        g1[1] = (int)((1024u & 0xFFFFu) << 16);       // tensor_dim0[15:0]
        g1[2] = (int)((1024u >> 16) | (1u << 16));    // tensor_dim0[31:16], tensor_dim1=1
        g1[3] = (int)(1024u << 16);                   // tile_dim0 = 1024
        g1[4] = 0;                                    // tile_dim1/2 unused
        g1[5] = 1024;                                 // tensor_dim0_stride
        g1[6] = 0;
        g1[7] = 0;
        i32x4 z4 = {0, 0, 0, 0};
        i32x8 z8 = {0, 0, 0, 0, 0, 0, 0, 0};
        __builtin_amdgcn_tensor_load_to_lds(g0, g1, z4, z4, z8, 0);
    };
#else
    // Fallback: immediate load->store (short-lived regs, no spill).
    auto directStage = [&](int g) {
        uint32_t* buf = bstage + (g & 1) * GRP_DWORDS;
        const uint4* gp = (const uint4*)(wp + (size_t)g * GRP_DWORDS) + tid * 4;
        uint4* lp = (uint4*)buf + tid * 4;
        #pragma unroll
        for (int i = 0; i < 4; ++i) lp[i] = gp[i];
    };
#endif

#if USE_TDM
    if (wave == 0) tdmStage(0);   // overlap first stage with feature phases
#endif

    // ---- Phase 1: radial basis (per tile: 16 edges x 8 radial) ----
    #pragma unroll
    for (int t = 0; t < TILES_PER_WAVE; ++t) {
        #pragma unroll
        for (int it = 0; it < 4; ++it) {
            int flat = it * 32 + lane;
            int e = flat >> 3, r = flat & 7;
            float d  = edge_length[eBase[t] + e] + 1e-8f;
            float kd = d * (float)(r + 1) * (C_PI / R_MAX_F);
            float bessel = __sinf(kd) / kd;
            float cut = (d < R_MAX_F) ? 0.5f * (__cosf(C_PI * d / R_MAX_F) + 1.0f) : 0.0f;
            TS[t]->rb[e][r] = bessel * cut;
        }
    }
    __syncthreads();

    // ---- Phase 2: h = silu(rb@fc_w1/sqrt(8))*SILU_C * 0.125 (1/sqrt(64)) ----
    #pragma unroll
    for (int t = 0; t < TILES_PER_WAVE; ++t) {
        for (int it = 0; it < 32; ++it) {
            int flat = it * 32 + lane;
            int e = flat >> 6, k = flat & 63;
            float acc = 0.0f;
            #pragma unroll
            for (int r = 0; r < 8; ++r) acc += TS[t]->rb[e][r] * fc_w1[r * 64 + k];
            acc *= INV_SQRT8;
            float h = (acc / (1.0f + __expf(-acc))) * (SILU_C * 0.125f);
            TS[t]->hbuf[e][k] = (_Float16)h;
        }
    }

    // ---- Phase 3: per-edge features (lanes 0..15, one edge each) ----
    #pragma unroll
    for (int t = 0; t < TILES_PER_WAVE; ++t) {
        if (lane < 16) {
            WaveShm& S = *TS[t];
            int e = lane;
            const float* ea = edge_attr + (size_t)(eBase[t] + e) * 9;
            float a0 = ea[0];
            float a1x = ea[1], a1y = ea[2], a1z = ea[3];
            float b0 = ea[4], b1 = ea[5], b2 = ea[6], b3 = ea[7], b4 = ea[8];
            int src = edge_src[eBase[t] + e];
            S.dstIdx[e] = edge_dst[eBase[t] + e];
            const float* xr = x + (size_t)src * DIM_IN;

            float cA = A_SC * a0;
            float cE = A_VE * INV3;
            #pragma unroll
            for (int u = 0; u < 32; ++u) {
                float s = xr[u];
                S.fA[u][e] = cA * s;
                S.fE[u][e] = cE * s;
            }
            // t[i][j] from a2 (CG121 hard-coded, perm = [1,2,0])
            float t00 = (-S6C * b2 - S2C * b4) * R5C;
            float t01 = ( S2C * b1) * R5C;
            float t02 = ( S2C * b0) * R5C;
            float t10 = ( S2C * b1) * R5C;
            float t11 = ( 2.0f * S6C * b2) * R5C;
            float t12 = ( S2C * b3) * R5C;
            float t20 = ( S2C * b0) * R5C;
            float t21 = ( S2C * b3) * R5C;
            float t22 = (-S6C * b2 + S2C * b4) * R5C;

            float cB = A_SC * INV3;
            float cF = A_VE * INV3 * a0;
            #pragma unroll
            for (int u = 0; u < 16; ++u) {
                float v0 = xr[32 + 3 * u + 0];
                float v1 = xr[32 + 3 * u + 1];
                float v2 = xr[32 + 3 * u + 2];
                float vd = v0 * a1x + v1 * a1y + v2 * a1z;
                S.fB[u][e]    = cB * vd;
                S.fF[u][0][e] = cF * v0;
                S.fF[u][1][e] = cF * v1;
                S.fF[u][2][e] = cF * v2;
                S.fG[u][0][e] = A_VE * (v0 * t00 + v1 * t10 + v2 * t20);
                S.fG[u][1][e] = A_VE * (v0 * t01 + v1 * t11 + v2 * t21);
                S.fG[u][2][e] = A_VE * (v0 * t02 + v1 * t12 + v2 * t22);
            }
            S.a1v[0][e] = a1x; S.a1v[1][e] = a1y; S.a1v[2][e] = a1z;
        }
    }
#if !USE_TDM
    directStage(0);
#endif
#if USE_TDM
    __builtin_amdgcn_s_wait_tensorcnt(0);
#endif
    __syncthreads();

    // ---- Phase 4: A operands (16x64 f16, K split 0-31 / 32-63), per tile ----
    union AOp { v16h v; v8h h[2]; } alo[TILES_PER_WAVE], ahi[TILES_PER_WAVE];
    {
        int m = lane & 15;
        int koff = (lane >> 4) * 8;
        #pragma unroll
        for (int t = 0; t < TILES_PER_WAVE; ++t) {
            const _Float16* hb = &TS[t]->hbuf[m][0];
            alo[t].h[0] = *(const v8h*)(hb + koff);
            alo[t].h[1] = *(const v8h*)(hb + 16 + koff);
            ahi[t].h[0] = *(const v8h*)(hb + 32 + koff);
            ahi[t].h[1] = *(const v8h*)(hb + 48 + koff);
        }
    }
    const int rowoff = (lane >> 4) * 8;   // D layout: vgpr r holds edge r or r+8

    Acc acc[TILES_PER_WAVE] = {};

    auto loadF8 = [&](const float* row, float* f) {
        float4 x0 = *(const float4*)(row + rowoff);
        float4 x1 = *(const float4*)(row + rowoff + 4);
        f[0]=x0.x; f[1]=x0.y; f[2]=x0.z; f[3]=x0.w;
        f[4]=x1.x; f[5]=x1.y; f[6]=x1.z; f[7]=x1.w;
    };
    auto accum8 = [&](float (&a)[8], v8f d, const float* f) {
        #pragma unroll
        for (int r = 0; r < 8; ++r) a[r] += d[r] * f[r];
    };
    // pipeline pieces (uniform across block)
    auto preStep = [&](int g) -> const uint32_t* {
#if USE_TDM
        if (g + 1 < NGROUP && wave == 0) tdmStage(g + 1);
#endif
        return bstage + (g & 1) * GRP_DWORDS;
    };
    auto postStep = [&](int g) {
#if !USE_TDM
        if (g + 1 < NGROUP) directStage(g + 1);
#endif
#if USE_TDM
        __builtin_amdgcn_s_wait_tensorcnt(0);
#endif
        __syncthreads();
    };
    union BO { v16h v; uint4 q[2]; };
    auto loadB = [&](const uint32_t* buf, int cc, BO& blo, BO& bhi) {
        const uint32_t* p0 = buf + ((cc * 2 + 0) * 32 + lane) * 8;
        const uint32_t* p1 = buf + ((cc * 2 + 1) * 32 + lane) * 8;
        blo.q[0] = *(const uint4*)p0;  blo.q[1] = *(const uint4*)(p0 + 4);
        bhi.q[0] = *(const uint4*)p1;  bhi.q[1] = *(const uint4*)(p1 + 4);
    };
    auto wmma2 = [&](int t, const BO& blo, const BO& bhi) -> v8f {
        v8f z = {};
        z = __builtin_amdgcn_wmma_f32_16x16x32_f16(false, alo[t].v, false, blo.v,
                                                   (short)0, z, false, false);
        z = __builtin_amdgcn_wmma_f32_16x16x32_f16(false, ahi[t].v, false, bhi.v,
                                                   (short)0, z, false, false);
        return z;
    };

    // ---- Main loops: statically dispatched per weight block ----
    // Block A: groups [0,16), chunks 0..63 -> scalars, feat fA
    for (int g = 0; g < 16; ++g) {
        const uint32_t* buf = preStep(g);
        #pragma unroll
        for (int cc = 0; cc < 4; ++cc) {
            BO blo, bhi; loadB(buf, cc, blo, bhi);
            int u = 2 * g + (cc >> 1);
            #pragma unroll
            for (int t = 0; t < TILES_PER_WAVE; ++t) {
                v8f z = wmma2(t, blo, bhi);
                float f[8]; loadF8(&TS[t]->fA[u][0], f);
                if (cc & 1) accum8(acc[t].sc1, z, f);
                else        accum8(acc[t].sc0, z, f);
            }
        }
        postStep(g);
    }
    // Block B: groups [16,24), chunks 64..95 -> scalars, feat fB
    for (int g = 16; g < 24; ++g) {
        const uint32_t* buf = preStep(g);
        #pragma unroll
        for (int cc = 0; cc < 4; ++cc) {
            BO blo, bhi; loadB(buf, cc, blo, bhi);
            int u = 2 * (g - 16) + (cc >> 1);
            #pragma unroll
            for (int t = 0; t < TILES_PER_WAVE; ++t) {
                v8f z = wmma2(t, blo, bhi);
                float f[8]; loadF8(&TS[t]->fB[u][0], f);
                if (cc & 1) accum8(acc[t].sc1, z, f);
                else        accum8(acc[t].sc0, z, f);
            }
        }
        postStep(g);
    }
    // Block C: groups [24,32), chunks 96..127 -> gates, feat fA
    for (int g = 24; g < 32; ++g) {
        const uint32_t* buf = preStep(g);
        #pragma unroll
        for (int cc = 0; cc < 4; ++cc) {
            BO blo, bhi; loadB(buf, cc, blo, bhi);
            int u = 4 * (g - 24) + cc;
            #pragma unroll
            for (int t = 0; t < TILES_PER_WAVE; ++t) {
                v8f z = wmma2(t, blo, bhi);
                float f[8]; loadF8(&TS[t]->fA[u][0], f);
                accum8(acc[t].gt, z, f);
            }
        }
        postStep(g);
    }
    // Block D: groups [32,36), chunks 128..143 -> gates, feat fB
    for (int g = 32; g < 36; ++g) {
        const uint32_t* buf = preStep(g);
        #pragma unroll
        for (int cc = 0; cc < 4; ++cc) {
            BO blo, bhi; loadB(buf, cc, blo, bhi);
            int u = 4 * (g - 32) + cc;
            #pragma unroll
            for (int t = 0; t < TILES_PER_WAVE; ++t) {
                v8f z = wmma2(t, blo, bhi);
                float f[8]; loadF8(&TS[t]->fB[u][0], f);
                accum8(acc[t].gt, z, f);
            }
        }
        postStep(g);
    }
    // Block E: groups [36,44), chunks 144..175 -> tmpE, feat fE
    for (int g = 36; g < 44; ++g) {
        const uint32_t* buf = preStep(g);
        #pragma unroll
        for (int cc = 0; cc < 4; ++cc) {
            BO blo, bhi; loadB(buf, cc, blo, bhi);
            int u = 4 * (g - 36) + cc;
            #pragma unroll
            for (int t = 0; t < TILES_PER_WAVE; ++t) {
                v8f z = wmma2(t, blo, bhi);
                float f[8]; loadF8(&TS[t]->fE[u][0], f);
                accum8(acc[t].te, z, f);
            }
        }
        postStep(g);
    }
    // Block F: groups [44,48), chunks 176..191 -> vecs, feat fF
    for (int g = 44; g < 48; ++g) {
        const uint32_t* buf = preStep(g);
        #pragma unroll
        for (int cc = 0; cc < 4; ++cc) {
            BO blo, bhi; loadB(buf, cc, blo, bhi);
            int u = 4 * (g - 44) + cc;
            #pragma unroll
            for (int t = 0; t < TILES_PER_WAVE; ++t) {
                v8f z = wmma2(t, blo, bhi);
                const float* base = &TS[t]->fF[u][0][0];
                float f0[8], f1[8], f2[8];
                loadF8(base, f0); loadF8(base + 16, f1); loadF8(base + 32, f2);
                accum8(acc[t].vc0, z, f0);
                accum8(acc[t].vc1, z, f1);
                accum8(acc[t].vc2, z, f2);
            }
        }
        postStep(g);
    }
    // Block G: groups [48,52), chunks 192..207 -> vecs, feat fG
    for (int g = 48; g < 52; ++g) {
        const uint32_t* buf = preStep(g);
        #pragma unroll
        for (int cc = 0; cc < 4; ++cc) {
            BO blo, bhi; loadB(buf, cc, blo, bhi);
            int u = 4 * (g - 48) + cc;
            #pragma unroll
            for (int t = 0; t < TILES_PER_WAVE; ++t) {
                v8f z = wmma2(t, blo, bhi);
                const float* base = &TS[t]->fG[u][0][0];
                float f0[8], f1[8], f2[8];
                loadF8(base, f0); loadF8(base + 16, f1); loadF8(base + 32, f2);
                accum8(acc[t].vc0, z, f0);
                accum8(acc[t].vc1, z, f1);
                accum8(acc[t].vc2, z, f2);
            }
        }
        postStep(g);
    }

    // ---- Epilogue: vecs += tmpE * a1[j]; scatter with f32 atomics ----
    const int n = lane & 15;
    #pragma unroll
    for (int t = 0; t < TILES_PER_WAVE; ++t) {
        WaveShm& S = *TS[t];
        Acc& A = acc[t];
        float g0[8], g1[8], g2[8];
        loadF8(&S.a1v[0][0], g0); loadF8(&S.a1v[1][0], g1); loadF8(&S.a1v[2][0], g2);
        #pragma unroll
        for (int r = 0; r < 8; ++r) {
            A.vc0[r] += A.te[r] * g0[r];
            A.vc1[r] += A.te[r] * g1[r];
            A.vc2[r] += A.te[r] * g2[r];
        }
        #pragma unroll
        for (int r = 0; r < 8; ++r) {
            int row = S.dstIdx[r + rowoff] * M_CH;
            atomicAdd(&m_sum[row + n],            A.sc0[r]);
            atomicAdd(&m_sum[row + 16 + n],       A.sc1[r]);
            atomicAdd(&m_sum[row + 32 + n],       A.gt[r]);
            atomicAdd(&m_sum[row + 48 + 3*n + 0], A.vc0[r]);
            atomicAdd(&m_sum[row + 48 + 3*n + 1], A.vc1[r]);
            atomicAdd(&m_sum[row + 48 + 3*n + 2], A.vc2[r]);
        }
        if (lane < 16) atomicAdd(&cnt[S.dstIdx[lane]], 1.0f);
    }
}

// ---------------------------------------------------------------------------
// Kernel 4: node update. One thread per (node, out-channel), 8000*80 threads.
// ---------------------------------------------------------------------------
__global__ void node_kernel(const float* __restrict__ x,
                            const float* __restrict__ sc_ws,
                            const float* __restrict__ sc_wv,
                            const float* __restrict__ m_sum,
                            const float* __restrict__ cnt,
                            float* __restrict__ out) {
    int id = blockIdx.x * 256 + threadIdx.x;
    if (id >= N_NODES * DIM_IN) return;
    int nde = id / DIM_IN;
    int ch  = id - nde * DIM_IN;
    float inv = 1.0f / fmaxf(cnt[nde], 1.0f);
    const float* ms = m_sum + (size_t)nde * M_CH;
    const float* xr = x + (size_t)nde * DIM_IN;
    float o;
    if (ch < 32) {
        float mi = ms[ch] * inv;
        float gs = (mi / (1.0f + __expf(-mi))) * SILU_C;
        float acc = 0.0f;
        #pragma unroll
        for (int u = 0; u < 32; ++u) acc += xr[u] * sc_ws[u * 32 + ch];
        o = gs + acc * INV_SQRT32;
    } else {
        int w = (ch - 32) / 3, j = (ch - 32) % 3;
        float gg = SIG_C / (1.0f + __expf(-ms[32 + w] * inv));
        float gv = ms[48 + 3 * w + j] * inv * gg;
        float acc = 0.0f;
        #pragma unroll
        for (int u = 0; u < 16; ++u) acc += xr[32 + 3 * u + j] * sc_wv[u * 16 + w];
        o = gv + acc * 0.25f;
    }
    out[id] = o;
}

// ---------------------------------------------------------------------------
extern "C" void kernel_launch(void* const* d_in, const int* in_sizes, int n_in,
                              void* d_out, int out_size, void* d_ws, size_t ws_size,
                              hipStream_t stream) {
    const float* x           = (const float*)d_in[0];
    const float* edge_attr   = (const float*)d_in[1];
    const float* edge_length = (const float*)d_in[2];
    const int*   edge_src    = (const int*)  d_in[3];
    const int*   edge_dst    = (const int*)  d_in[4];
    const float* fc_w1       = (const float*)d_in[5];
    const float* fc_w2       = (const float*)d_in[6];
    const float* sc_ws       = (const float*)d_in[7];
    const float* sc_wv       = (const float*)d_in[8];
    float* out = (float*)d_out;

    // ws: [packed f16 W2 | m_sum | cnt]  ~3.4 MB total
    uint32_t* packed = (uint32_t*)d_ws;
    float* m_sum = (float*)((char*)d_ws + (size_t)PACK_DWORDS * 4);
    float* cnt   = m_sum + MSUM_FLOATS;

    // 1) pack fc_w2 into WMMA B-operand f16 layout (106496 dwords)
    pack_w2_kernel<<<PACK_DWORDS / 256, 256, 0, stream>>>(fc_w2, packed);
    // 2) zero accumulators (m_sum + cnt contiguous)
    {
        int nz = MSUM_FLOATS + CNT_FLOATS;
        zero_kernel<<<(nz + 255) / 256, 256, 0, stream>>>(m_sum, nz);
    }
    // 3) fused edge kernel: 5000 tiles = 625 blocks x (4 waves x 2 tiles)
    //    129 KB dynamic LDS: 8 per-tile feature regions + double-buffered
    //    B staging window (fits 320 KB/WGP; 2 blocks per WGP)
    edge_kernel<<<625, 128, DYN_LDS_BYTES, stream>>>(x, edge_attr, edge_length,
                                                     edge_src, edge_dst, fc_w1,
                                                     packed, m_sum, cnt);
    // 4) node update: 640000 threads
    node_kernel<<<(N_NODES * DIM_IN) / 256, 256, 0, stream>>>(
        x, sc_ws, sc_wv, m_sum, cnt, out);
}
